// ModifiedBertAttention_23776938951214
// MI455X (gfx1250) — matrix operand order repaired and checked by reading, not compiled, not run on hardware
//
#include <hip/hip_runtime.h>
#include <hip/hip_bf16.h>
#include <stdint.h>

// ---------------- problem constants (B=2, L=2048, C=1024, H=16, DH=64) ----
static constexpr int Bk  = 2;
static constexpr int Lk  = 2048;
static constexpr int Ck  = 1024;
static constexpr int Hk  = 16;
static constexpr int DHk = 64;
static constexpr float LN_EPS = 1e-12f;

typedef __attribute__((ext_vector_type(16))) __bf16 v16bf;
typedef __attribute__((ext_vector_type(8)))  __bf16 v8bf;
typedef __attribute__((ext_vector_type(4)))  __bf16 v4bf;
typedef __attribute__((ext_vector_type(8)))  float  v8f;

// ---------------- WMMA fragment helpers (contiguous k order, same for A and B) --------
// A fragment 16x32 bf16: lane holds row M=lane&15; kh=lane>>4;
//   K = 16*kh .. 16*kh+15 contiguous.
__device__ __forceinline__ v16bf ld_frag_a(const __bf16* base, int stride, int lane) {
    const __bf16* p = base + (size_t)(lane & 15) * stride + ((lane >> 4) << 4);
    v8bf lo = *(const v8bf*)(p);
    v8bf hi = *(const v8bf*)(p + 8);
    return __builtin_shufflevector(lo, hi, 0,1,2,3,4,5,6,7,8,9,10,11,12,13,14,15);
}
// B fragment 32x16 bf16 fed from memory laid out [N][K] row-major:
//   lane holds column N=lane&15; kh=lane>>4; K = 16*kh .. 16*kh+15 contiguous.
__device__ __forceinline__ v16bf ld_frag_b(const __bf16* base, int stride, int lane) {
    const __bf16* p = base + (size_t)(lane & 15) * stride + ((lane >> 4) << 4);
    v8bf lo = *(const v8bf*)(p);
    v8bf hi = *(const v8bf*)(p + 8);
    return __builtin_shufflevector(lo, hi, 0,1,2,3,4,5,6,7,8,9,10,11,12,13,14,15);
}
__device__ __forceinline__ v8f wmma_bf16(v16bf a, v16bf b, v8f c) {
    return __builtin_amdgcn_wmma_f32_16x16x32_bf16(false, a, false, b, (short)0, c, false, false);
}

// ---------------- K1: x (f32) -> xb (bf16) --------------------------------
__global__ __launch_bounds__(256) void k_cvt_x(const float* __restrict__ x,
                                              __bf16* __restrict__ xb) {
    size_t i = ((size_t)blockIdx.x * 256 + threadIdx.x) * 4;
    float4 v = *(const float4*)(x + i);
    v4bf o; o[0] = (__bf16)v.x; o[1] = (__bf16)v.y; o[2] = (__bf16)v.z; o[3] = (__bf16)v.w;
    *(v4bf*)(xb + i) = o;
}

// ---------------- K0: transpose 1024x1024 f32 -> bf16 [n][k] --------------
// z = 0,1,2 : Wq/Wk/Wv [c][n] -> Wt[n][c];  z = 3 : Wo [k][co] -> WoT[co][k]
__global__ __launch_bounds__(256) void k_wtrans(const float* __restrict__ Wq,
                                               const float* __restrict__ Wk,
                                               const float* __restrict__ Wv,
                                               const float* __restrict__ Wo,
                                               __bf16* __restrict__ dst_base) {
    __shared__ float t[32][33];
    int z = blockIdx.z;
    const float* src = (z == 0) ? Wq : (z == 1) ? Wk : (z == 2) ? Wv : Wo;
    __bf16* dst = dst_base + (size_t)z * Ck * Ck;
    int tx = threadIdx.x, ty = threadIdx.y;
    int r0 = blockIdx.y * 32, c0 = blockIdx.x * 32;
#pragma unroll
    for (int i = 0; i < 4; ++i)
        t[ty + 8 * i][tx] = src[(size_t)(r0 + ty + 8 * i) * Ck + c0 + tx];
    __syncthreads();
#pragma unroll
    for (int i = 0; i < 4; ++i)
        dst[(size_t)(c0 + ty + 8 * i) * Ck + r0 + tx] = (__bf16)t[tx][ty + 8 * i];
}

// ---------------- K2: QKV projection GEMM (WMMA bf16) ---------------------
// grid: (L/64, C/128, B*3); block 256 = 8 waves; wave computes 32(l) x 32(n).
// Q gets bias then *0.125 (1/sqrt(DH)); V stored transposed Vt[b,h][d][l].
__global__ __launch_bounds__(256) void k_qkv(const __bf16* __restrict__ xb,
                                             const __bf16* __restrict__ Wt3,
                                             const float* __restrict__ bq,
                                             const float* __restrict__ bk,
                                             const float* __restrict__ bv,
                                             __bf16* __restrict__ Qb,
                                             __bf16* __restrict__ Kb,
                                             __bf16* __restrict__ Vt) {
    int z = blockIdx.z, b = z / 3, mat = z % 3;
    int tid = threadIdx.x, lane = tid & 31, w = tid >> 5;
    int l0 = blockIdx.x * 64 + (w >> 2) * 32;
    int n0 = blockIdx.y * 128 + (w & 3) * 32;
    const __bf16* wt   = Wt3 + (size_t)mat * Ck * Ck;
    const __bf16* xrow = xb + (size_t)b * Lk * Ck;

    v8f acc[2][2] = {};
    for (int kk = 0; kk < Ck; kk += 32) {
        v16bf a0 = ld_frag_a(xrow + (size_t)l0 * Ck + kk, Ck, lane);
        v16bf a1 = ld_frag_a(xrow + (size_t)(l0 + 16) * Ck + kk, Ck, lane);
        v16bf b0 = ld_frag_b(wt + (size_t)n0 * Ck + kk, Ck, lane);
        v16bf b1 = ld_frag_b(wt + (size_t)(n0 + 16) * Ck + kk, Ck, lane);
        acc[0][0] = wmma_bf16(a0, b0, acc[0][0]);
        acc[0][1] = wmma_bf16(a0, b1, acc[0][1]);
        acc[1][0] = wmma_bf16(a1, b0, acc[1][0]);
        acc[1][1] = wmma_bf16(a1, b1, acc[1][1]);
    }
    const float* bias = (mat == 0) ? bq : (mat == 1) ? bk : bv;
    int nlo = lane & 15, g = lane >> 4;
#pragma unroll
    for (int j = 0; j < 2; ++j) {
        int n = n0 + 16 * j + nlo;
        float bv_ = bias[n];
        int hh = n >> 6, d = n & 63;
#pragma unroll
        for (int i = 0; i < 2; ++i) {
#pragma unroll
            for (int r = 0; r < 8; ++r) {
                int m = l0 + 16 * i + r + 8 * g;   // global query index l
                float val = acc[i][j][r] + bv_;
                if (mat == 0) {
                    Qb[((size_t)(b * Hk + hh) * Lk + m) * DHk + d] = (__bf16)(val * 0.125f);
                } else if (mat == 1) {
                    Kb[((size_t)(b * Hk + hh) * Lk + m) * DHk + d] = (__bf16)val;
                } else {
                    Vt[((size_t)(b * Hk + hh) * DHk + d) * Lk + m] = (__bf16)val;
                }
            }
        }
    }
}

// ---------------- K3: flash attention per (b,h,16-row q-tile) -------------
// Computes S^T = K * Q^T so every lane owns one query row's softmax stats;
// exp(S^T) is repacked in registers into the B-fragment of ctx^T = Vt * P.
__global__ __launch_bounds__(128) void k_attn(const __bf16* __restrict__ Qb,
                                              const __bf16* __restrict__ Kb,
                                              const __bf16* __restrict__ Vt,
                                              const float* __restrict__ mask,
                                              __bf16* __restrict__ ctxb) {
    int lane = threadIdx.x & 31;
    int wid  = blockIdx.x * 4 + (threadIdx.x >> 5);
    int lt = wid & (Lk / 16 - 1);
    int h  = (wid >> 7) & (Hk - 1);
    int b  = wid >> 11;
    int l0 = lt * 16;
    int g  = lane >> 4;

    const __bf16* Qh = Qb + (size_t)(b * Hk + h) * Lk * DHk;
    const __bf16* Kh = Kb + (size_t)(b * Hk + h) * Lk * DHk;
    const __bf16* Vh = Vt + (size_t)(b * Hk + h) * DHk * Lk;
    const float* mrow = mask + (size_t)b * Lk * Lk + (size_t)(l0 + (lane & 15)) * Lk;

    // Q rows l0..l0+15 as B-fragments (K = d: 0..31 and 32..63)
    v16bf qf0 = ld_frag_b(Qh + (size_t)l0 * DHk + 0,  DHk, lane);
    v16bf qf1 = ld_frag_b(Qh + (size_t)l0 * DHk + 32, DHk, lane);

    v8f cacc[4] = {};
    float Mrow = -3.0e38f, Srow = 0.0f;
    const v8f vz = {};

    for (int m0 = 0; m0 < Lk; m0 += 32) {
        // K-matrix A-fragments: tile0 rows m0..m0+15, tile1 rows m0+16..m0+31
        v16bf k00 = ld_frag_a(Kh + (size_t)m0 * DHk + 0,  DHk, lane);
        v16bf k01 = ld_frag_a(Kh + (size_t)m0 * DHk + 32, DHk, lane);
        v16bf k10 = ld_frag_a(Kh + (size_t)(m0 + 16) * DHk + 0,  DHk, lane);
        v16bf k11 = ld_frag_a(Kh + (size_t)(m0 + 16) * DHk + 32, DHk, lane);
        v8f s0 = wmma_bf16(k00, qf0, vz); s0 = wmma_bf16(k01, qf1, s0);
        v8f s1 = wmma_bf16(k10, qf0, vz); s1 = wmma_bf16(k11, qf1, s1);

        // additive mask: reg r of tile0 is m = m0 + r + 8g (contiguous per lane)
        const float* mp = mrow + m0 + 8 * g;
        float4 ma = *(const float4*)(mp);      float4 mb_ = *(const float4*)(mp + 4);
        float4 mc = *(const float4*)(mp + 16); float4 md  = *(const float4*)(mp + 20);
        s0[0] += ma.x; s0[1] += ma.y; s0[2] += ma.z; s0[3] += ma.w;
        s0[4] += mb_.x; s0[5] += mb_.y; s0[6] += mb_.z; s0[7] += mb_.w;
        s1[0] += mc.x; s1[1] += mc.y; s1[2] += mc.z; s1[3] += mc.w;
        s1[4] += md.x; s1[5] += md.y; s1[6] += md.z; s1[7] += md.w;

        // online softmax for column l = lane&15 (lane and lane^16 share l)
        float mloc = s0[0];
#pragma unroll
        for (int r = 1; r < 8; ++r) mloc = fmaxf(mloc, s0[r]);
#pragma unroll
        for (int r = 0; r < 8; ++r) mloc = fmaxf(mloc, s1[r]);
        mloc = fmaxf(mloc, __shfl_xor(mloc, 16, 32));
        float Mn = fmaxf(Mrow, mloc);
        float corr = __expf(Mrow - Mn);
        float p0[8], p1[8], sl = 0.0f;
#pragma unroll
        for (int r = 0; r < 8; ++r) {
            p0[r] = __expf(s0[r] - Mn);
            p1[r] = __expf(s1[r] - Mn);
            sl += p0[r] + p1[r];
        }
        sl += __shfl_xor(sl, 16, 32);
        Srow = Srow * corr + sl;
        Mrow = Mn;
#pragma unroll
        for (int t = 0; t < 4; ++t)
#pragma unroll
            for (int r = 0; r < 8; ++r) cacc[t][r] *= corr;

        // repack exp(S^T) into B-fragment of ctx matmul (K = m_local 0..31)
        v16bf pf;
#pragma unroll
        for (int r = 0; r < 8; ++r) {
            float e0 = __shfl_xor(p0[r], 16, 32);
            float e1 = __shfl_xor(p1[r], 16, 32);
            float lo = g ? e1 : p0[r];
            float hi = g ? p1[r] : e0;
            pf[r]     = (__bf16)lo;
            pf[8 + r] = (__bf16)hi;
        }
        // ctx^T[d][l] += Vt(A, rows d) * P(B)
#pragma unroll
        for (int t = 0; t < 4; ++t) {
            v16bf vf = ld_frag_a(Vh + (size_t)(16 * t) * Lk + m0, Lk, lane);
            cacc[t] = wmma_bf16(vf, pf, cacc[t]);
        }
    }
    float inv = 1.0f / Srow;
    __bf16* crow = ctxb + ((size_t)b * Lk + l0 + (lane & 15)) * Ck + h * DHk;
#pragma unroll
    for (int t = 0; t < 4; ++t)
#pragma unroll
        for (int r = 0; r < 8; ++r)
            crow[16 * t + r + 8 * g] = (__bf16)(cacc[t][r] * inv);
}

// ---------------- K4: out-proj + residual + LayerNorm ---------------------
// grid: B*L/16 blocks of 256; wave w covers n in [w*128, w*128+128).
__global__ __launch_bounds__(256) void k_oproj_ln(const __bf16* __restrict__ ctxb,
                                                  const __bf16* __restrict__ WoT,
                                                  const float* __restrict__ bo,
                                                  const float* __restrict__ x,
                                                  const float* __restrict__ gamma,
                                                  const float* __restrict__ beta,
                                                  float* __restrict__ out) {
    __shared__ float yt[16 * Ck];   // 64 KB
    int tid = threadIdx.x, lane = tid & 31, w = tid >> 5;
    size_t R0 = (size_t)blockIdx.x * 16;
    int n0 = w * 128;
    const __bf16* arow = ctxb + R0 * Ck;

    v8f acc[8] = {};
    for (int kk = 0; kk < Ck; kk += 32) {
        v16bf a = ld_frag_a(arow + kk, Ck, lane);
#pragma unroll
        for (int j = 0; j < 8; ++j) {
            v16bf bf = ld_frag_b(WoT + (size_t)(n0 + 16 * j) * Ck + kk, Ck, lane);
            acc[j] = wmma_bf16(a, bf, acc[j]);
        }
    }
    int nlo = lane & 15, g = lane >> 4;
#pragma unroll
    for (int j = 0; j < 8; ++j) {
        int n = n0 + 16 * j + nlo;
        float bb = bo[n];
#pragma unroll
        for (int r = 0; r < 8; ++r) {
            int m = r + 8 * g;
            yt[m * Ck + n] = acc[j][r] + bb + x[(R0 + m) * Ck + n];
        }
    }
    __syncthreads();

    int row = tid >> 4, sub = tid & 15;     // 16 threads per row
    float s = 0.0f, s2 = 0.0f;
#pragma unroll 8
    for (int j = 0; j < 64; ++j) {
        float v = yt[row * Ck + sub + 16 * j];
        s += v; s2 += v * v;
    }
#pragma unroll
    for (int m = 8; m >= 1; m >>= 1) {      // xor masks stay within 16-lane half
        s  += __shfl_xor(s,  m, 32);
        s2 += __shfl_xor(s2, m, 32);
    }
    float mu = s * (1.0f / Ck);
    float var = s2 * (1.0f / Ck) - mu * mu;
    float rstd = rsqrtf(var + LN_EPS);
    size_t R = R0 + row;
#pragma unroll 8
    for (int j = 0; j < 64; ++j) {
        int c = sub + 16 * j;
        float v = yt[row * Ck + c];
        out[R * Ck + c] = (v - mu) * rstd * gamma[c] + beta[c];
    }
}

// ---------------- host launcher -------------------------------------------
extern "C" void kernel_launch(void* const* d_in, const int* in_sizes, int n_in,
                              void* d_out, int out_size, void* d_ws, size_t ws_size,
                              hipStream_t stream) {
    const float* x     = (const float*)d_in[0];
    const float* mask  = (const float*)d_in[1];
    const float* Wq    = (const float*)d_in[2];
    const float* bq    = (const float*)d_in[3];
    const float* Wk    = (const float*)d_in[4];
    const float* bk    = (const float*)d_in[5];
    const float* Wv    = (const float*)d_in[6];
    const float* bv    = (const float*)d_in[7];
    const float* Wo    = (const float*)d_in[8];
    const float* bo    = (const float*)d_in[9];
    const float* gamma = (const float*)d_in[10];
    const float* beta  = (const float*)d_in[11];
    float* out = (float*)d_out;

    uint8_t* ws = (uint8_t*)d_ws;
    const size_t MB = 1024 * 1024;
    __bf16* xb   = (__bf16*)(ws + 0);          //  8 MB: [B][L][C] bf16
    __bf16* Wt3  = (__bf16*)(ws + 8  * MB);    //  8 MB: WtQ,WtK,WtV,WoT  [n][k]
    __bf16* WoT  = (__bf16*)(ws + 14 * MB);    //  (4th transpose slot)
    __bf16* Qb   = (__bf16*)(ws + 16 * MB);    //  8 MB: [B][H][L][DH] (pre-scaled)
    __bf16* Kb   = (__bf16*)(ws + 24 * MB);    //  8 MB: [B][H][L][DH]
    __bf16* Vt   = (__bf16*)(ws + 32 * MB);    //  8 MB: [B][H][DH][L]
    __bf16* ctxb = (__bf16*)(ws + 40 * MB);    //  8 MB: [B][L][C]

    // 1) convert x to bf16
    k_cvt_x<<<(Bk * Lk * Ck) / (256 * 4), 256, 0, stream>>>(x, xb);
    // 2) transpose+convert the four 1024x1024 weight matrices
    k_wtrans<<<dim3(Ck / 32, Ck / 32, 4), dim3(32, 8), 0, stream>>>(Wq, Wk, Wv, Wo, Wt3);
    // 3) Q/K/V projections (WMMA), bias + Q scale folded, V transposed
    k_qkv<<<dim3(Lk / 64, Ck / 128, Bk * 3), 256, 0, stream>>>(
        xb, Wt3, bq, bk, bv, Qb, Kb, Vt);
    // 4) flash attention
    k_attn<<<(Bk * Hk * (Lk / 16)) / 4, 128, 0, stream>>>(Qb, Kb, Vt, mask, ctxb);
    // 5) output projection + residual + LayerNorm
    k_oproj_ln<<<(Bk * Lk) / 16, 256, 0, stream>>>(ctxb, WoT, bo, x, gamma, beta, out);
}
